// _Mutilscal_MHSA_75453985457234
// MI455X (gfx1250) — compile-verified, matrix operand order checked by reading, not run on hardware
//
#include <hip/hip_runtime.h>
#include <math.h>

// ---------------- types / helpers ----------------
typedef __attribute__((ext_vector_type(16))) __bf16 bf16x16;
typedef __attribute__((ext_vector_type(8)))  float  f32x8;

#define BSZ   8
#define CDIM  512
#define HWN   4096          // 64*64
#define NHEAD 8
#define HD    64
#define NPOOL 256           // 16*16
#define ATT_SCALE 0.125f    // 64^-0.5

__device__ __forceinline__ f32x8 wmma_bf16(bf16x16 a, bf16x16 b, f32x8 c) {
  // v_wmma_f32_16x16x32_bf16  (neg_a, A, neg_b, B, c_mod, C, reuse_a, reuse_b)
  return __builtin_amdgcn_wmma_f32_16x16x32_bf16(false, a, false, b, (short)0, c,
                                                 false, false);
}

__device__ __forceinline__ float silu_f(float v) { return v / (1.f + __expf(-v)); }

// Load a 16x32 bf16 fragment (A-layout; identical packing used for B via
// N-major LDS storage).  ISA lane layout: lanes 0-15 hold k {0..7,16..23},
// lanes 16-31 hold k {8..15,24..31}; row = lane & 15.
__device__ __forceinline__ bf16x16 ld_frag(const __bf16* base, int ld) {
  int lane = threadIdx.x & 31;
  const __bf16* p = base + (lane & 15) * ld + ((lane >> 4) << 3);
  bf16x16 r;
#pragma unroll
  for (int i = 0; i < 8; ++i) r[i] = p[i];
#pragma unroll
  for (int i = 0; i < 8; ++i) r[i + 8] = p[i + 16];
  return r;
}

// ---------------- elementwise converts ----------------
__global__ __launch_bounds__(256) void cvt_f2b(const float* __restrict__ in,
                                               __bf16* __restrict__ out, int n) {
  int i = blockIdx.x * 256 + threadIdx.x;
  if (i < n) out[i] = (__bf16)in[i];
}

// ---------------- SE path ----------------
__global__ __launch_bounds__(256) void gap_kernel(const float* __restrict__ x,
                                                  float* __restrict__ gap) {
  __shared__ float sm[256];
  int bc = blockIdx.x;                       // b*512 + c
  const float* p = x + (size_t)bc * HWN;
  float s = 0.f;
  for (int i = threadIdx.x; i < HWN; i += 256) s += p[i];
  sm[threadIdx.x] = s;
  __syncthreads();
  for (int st = 128; st > 0; st >>= 1) {
    if (threadIdx.x < st) sm[threadIdx.x] += sm[threadIdx.x + st];
    __syncthreads();
  }
  if (threadIdx.x == 0) gap[bc] = sm[0] * (1.f / (float)HWN);
}

__global__ __launch_bounds__(128) void se_fc1(const float* __restrict__ gap,
                                              const float* __restrict__ w,
                                              const float* __restrict__ bias,
                                              float* __restrict__ y) {
  int b = blockIdx.x, j = threadIdx.x;       // 128 outputs
  const float* g = gap + b * CDIM;
  const float* wr = w + j * CDIM;
  float s = bias[j];
  for (int c = 0; c < CDIM; ++c) s += g[c] * wr[c];
  y[b * 128 + j] = fminf(fmaxf(s, 0.f), 6.f);        // ReLU6
}

__global__ __launch_bounds__(512) void se_fc2(const float* __restrict__ y,
                                              const float* __restrict__ w,
                                              const float* __restrict__ bias,
                                              float* __restrict__ gate) {
  int b = blockIdx.x, c = threadIdx.x;       // 512 outputs
  const float* yr = y + b * 128;
  const float* wr = w + c * 128;
  float s = bias[c];
  for (int j = 0; j < 128; ++j) s += yr[j] * wr[j];
  gate[b * CDIM + c] = 1.f / (1.f + __expf(-s));
}

// ---------------- depthwise dilated 3x3 + scale/bias + SiLU ----------------
__global__ __launch_bounds__(256) void dwconv_silu(const __bf16* __restrict__ in,
                                                   const float* __restrict__ w,
                                                   const float* __restrict__ s,
                                                   const float* __restrict__ bi,
                                                   __bf16* __restrict__ out, int dil) {
  int idx = blockIdx.x * 256 + threadIdx.x;  // 8*128*4096
  int xw = idx & 63, yy = (idx >> 6) & 63, c = (idx >> 12) & 127, b = idx >> 19;
  const __bf16* pin = in + ((size_t)b * 128 + c) * HWN;
  const float* wc = w + c * 9;
  float acc = 0.f;
#pragma unroll
  for (int i = 0; i < 3; ++i) {
    int yi = yy + dil * (i - 1);
    if (yi < 0 || yi > 63) continue;
#pragma unroll
    for (int j = 0; j < 3; ++j) {
      int xi = xw + dil * (j - 1);
      if (xi < 0 || xi > 63) continue;
      acc += wc[i * 3 + j] * (float)pin[yi * 64 + xi];
    }
  }
  float v = silu_f(acc * s[c] + bi[c]);
  out[((size_t)b * 128 + c) * HWN + yy * 64 + xw] = (__bf16)v;
}

// ---------------- 4x4 block-mean pooling (64x64 -> 16x16), f32 -> bf16 ------
__global__ __launch_bounds__(256) void pool_kernel(const float* __restrict__ attn,
                                                   __bf16* __restrict__ pooled) {
  int idx = blockIdx.x * 256 + threadIdx.x;  // 8*512*256
  int px = idx & 15, py = (idx >> 4) & 15, c = (idx >> 8) & 511, b = idx >> 17;
  const float* p = attn + ((size_t)b * CDIM + c) * HWN;
  float s = 0.f;
#pragma unroll
  for (int i = 0; i < 4; ++i)
#pragma unroll
    for (int j = 0; j < 4; ++j) s += p[(py * 4 + i) * 64 + px * 4 + j];
  pooled[((size_t)b * CDIM + c) * NPOOL + py * 16 + px] = (__bf16)(s * (1.f / 16.f));
}

// ---------------- generic WMMA GEMM: D = epi(scale*(A@B)+bias) --------------
// A: [M,K] bf16 row-major (weights, no batch)
// B: [K,N] bf16 (batch stride K*N)
// Block: 256 thr = 8 waves (4 in M x 2 in N), tile 64x128, K-step 32.
enum {
  EPI_WRITE_BF16 = 1,
  EPI_ACC_F32    = 2,
  EPI_WRITE_F32  = 4,
  EPI_MULX       = 8,
  EPI_FIRST      = 16
};

template <int MODE>
__global__ __launch_bounds__(256)
void gemm_sb_silu(const __bf16* __restrict__ A, const __bf16* __restrict__ B,
                  const float* __restrict__ scale, const float* __restrict__ bias,
                  const float* __restrict__ xmul, float* __restrict__ outF,
                  __bf16* __restrict__ outB, int M, int K, int N) {
  __shared__ __align__(16) __bf16 As[64 * 32];   // [mLocal][k]
  __shared__ __align__(16) __bf16 Bs[128 * 32];  // [nLocal][k]  (transposed)

  int m0 = blockIdx.y * 64, n0 = blockIdx.x * 128, b = blockIdx.z;
  const __bf16* Bb = B + (size_t)b * K * N;
  int wave = threadIdx.x >> 5;
  int wm = wave & 3, wn = wave >> 2;

  f32x8 acc[4] = {};

  for (int k0 = 0; k0 < K; k0 += 32) {
    {  // stage A: 64 rows x 32 k, 8 bf16 per thread (b128)
      int row = threadIdx.x >> 2, ch = threadIdx.x & 3;
      *(uint4*)(&As[row * 32 + ch * 8]) =
          *(const uint4*)(A + (size_t)(m0 + row) * K + k0 + ch * 8);
    }
    {  // stage B transposed: [nLocal][k]
      int nl = threadIdx.x & 127, kh = (threadIdx.x >> 7) * 16;
#pragma unroll
      for (int kk = 0; kk < 16; ++kk)
        Bs[nl * 32 + kh + kk] = Bb[(size_t)(k0 + kh + kk) * N + n0 + nl];
      if (k0 + 32 < K)  // hint next B panel toward L2 (global_prefetch_b8)
        __builtin_prefetch(&Bb[(size_t)(k0 + 32 + kh) * N + n0 + nl], 0, 0);
    }
    __syncthreads();

    bf16x16 af = ld_frag(&As[(wm * 16) * 32], 32);
#pragma unroll
    for (int j = 0; j < 4; ++j) {
      bf16x16 bf = ld_frag(&Bs[(wn * 64 + j * 16) * 32], 32);
      acc[j] = wmma_bf16(af, bf, acc[j]);
    }
    __syncthreads();
  }

  // epilogue — C layout: VGPR r, lanes 0-15 -> M=r, lanes 16-31 -> M=r+8, N=lane&15
  int lane = threadIdx.x & 31;
  int half = lane >> 4, col = lane & 15;
#pragma unroll
  for (int j = 0; j < 4; ++j) {
    int n = n0 + wn * 64 + j * 16 + col;
#pragma unroll
    for (int r = 0; r < 8; ++r) {
      int m = m0 + wm * 16 + r + half * 8;
      float v = silu_f(acc[j][r] * scale[m] + bias[m]);
      size_t oi = ((size_t)b * M + m) * N + n;
      if (MODE & EPI_MULX) v *= xmul[oi];
      if (MODE & EPI_ACC_F32) outF[oi] = (MODE & EPI_FIRST) ? v : (outF[oi] + v);
      if (MODE & EPI_WRITE_F32) outF[oi] = v;
      if (MODE & EPI_WRITE_BF16) outB[oi] = (__bf16)v;
    }
  }
}

// ---------------- fused cross-attention + (gate*x) epilogue -----------------
// grid: (64 q-blocks of 64 rows, 8 heads, 8 batch); 4 waves, 16 q rows/wave.
// Online softmax over n1=256 keys in two 128-key LDS-resident chunks.
// All 8 S tiles of a chunk are computed first (16 back-to-back WMMA), then a
// SINGLE cross-lane reduction per row (max + sum) handles the whole chunk:
// 8x fewer ds_bpermute ops than per-tile reductions.
#define AW 4
__global__ __launch_bounds__(128)
void attention_kernel(const __bf16* __restrict__ xb, const __bf16* __restrict__ kv,
                      const float* __restrict__ x, const float* __restrict__ gate,
                      float* __restrict__ out) {
  __shared__ __align__(16) __bf16 Kl[128 * 64];        // [mLocal][d]  16KB
  __shared__ __align__(16) __bf16 Vl[64 * 128];        // [d][mLocal]  16KB
  __shared__ __align__(16) __bf16 Pl[AW][16 * 128];    // P chunk/wave 16KB

  int b = blockIdx.z, h = blockIdx.y;
  int wave = threadIdx.x >> 5, lane = threadIdx.x & 31;
  int q0 = blockIdx.x * (16 * AW) + wave * 16;
  int half = lane >> 4, col = lane & 15;

  const __bf16* xbh = xb + ((size_t)b * CDIM + h * HD) * HWN;
  const __bf16* kb  = kv + ((size_t)b * 1024 + h * HD) * NPOOL;
  const __bf16* vb  = kv + ((size_t)b * 1024 + 512 + h * HD) * NPOOL;

  // Q fragments: q[n][d] = x[b, h*64+d, n]  (strided gather from bf16 x)
  bf16x16 qf[2];
  {
    int row = lane & 15, kb8 = (lane >> 4) << 3;
#pragma unroll
    for (int kt = 0; kt < 2; ++kt)
#pragma unroll
      for (int i = 0; i < 16; ++i) {
        int d = kt * 32 + kb8 + (i < 8 ? i : i + 8);
        qf[kt][i] = xbh[(size_t)d * HWN + q0 + row];
      }
  }

  f32x8 o[4] = {};
  float mrun[8], lrun[8];
#pragma unroll
  for (int r = 0; r < 8; ++r) { mrun[r] = -1e30f; lrun[r] = 0.f; }

  for (int mc = 0; mc < 2; ++mc) {        // two 128-key chunks
    __syncthreads();
    {  // stage K (transposed to [m][d]) and V (as [d][m]); 128 threads
      int ml = threadIdx.x;               // 0..127
      for (int dd = 0; dd < 64; ++dd)
        Kl[ml * 64 + dd] = kb[(size_t)dd * NPOOL + mc * 128 + ml];
      int dl = threadIdx.x & 63, mh = (threadIdx.x >> 6) * 64;
      for (int mm = 0; mm < 64; ++mm)
        Vl[dl * 128 + mh + mm] = vb[(size_t)dl * NPOOL + mc * 128 + mh + mm];
    }
    __syncthreads();

    // ---- all 8 S tiles of this chunk (16 WMMA, pre-scaled) ----
    f32x8 st[8];
#pragma unroll
    for (int mt = 0; mt < 8; ++mt) {
      f32x8 sacc = {};
#pragma unroll
      for (int kt = 0; kt < 2; ++kt) {
        bf16x16 bf = ld_frag(&Kl[(mt * 16) * 64 + kt * 32], 64);
        sacc = wmma_bf16(qf[kt], bf, sacc);
      }
#pragma unroll
      for (int r = 0; r < 8; ++r) st[mt][r] = sacc[r] * ATT_SCALE;
    }

    // ---- chunk-wide online softmax: one lane-reduction per row ----
#pragma unroll
    for (int r = 0; r < 8; ++r) {
      float vm = st[0][r];
#pragma unroll
      for (int mt = 1; mt < 8; ++mt) vm = fmaxf(vm, st[mt][r]);
      vm = fmaxf(vm, __shfl_xor(vm, 1, 32));
      vm = fmaxf(vm, __shfl_xor(vm, 2, 32));
      vm = fmaxf(vm, __shfl_xor(vm, 4, 32));
      vm = fmaxf(vm, __shfl_xor(vm, 8, 32));
      float mn = fmaxf(mrun[r], vm);
      float corr = __expf(mrun[r] - mn);
      o[0][r] *= corr; o[1][r] *= corr; o[2][r] *= corr; o[3][r] *= corr;
      float ps = 0.f;
#pragma unroll
      for (int mt = 0; mt < 8; ++mt) {
        float pe = __expf(st[mt][r] - mn);
        ps += pe;
        Pl[wave][(r + 8 * half) * 128 + mt * 16 + col] = (__bf16)pe;
      }
      ps += __shfl_xor(ps, 1, 32);
      ps += __shfl_xor(ps, 2, 32);
      ps += __shfl_xor(ps, 4, 32);
      ps += __shfl_xor(ps, 8, 32);
      lrun[r] = lrun[r] * corr + ps;
      mrun[r] = mn;
    }
    __syncthreads();   // make wave's P tile LDS-visible before fragment reads

    // ---- P @ V: 4 chunks of 32 keys, 16 WMMA ----
#pragma unroll
    for (int ch = 0; ch < 4; ++ch) {
      bf16x16 af = ld_frag(&Pl[wave][ch * 32], 128);
#pragma unroll
      for (int dt = 0; dt < 4; ++dt) {
        bf16x16 bf = ld_frag(&Vl[(dt * 16) * 128 + ch * 32], 128);
        o[dt] = wmma_bf16(af, bf, o[dt]);
      }
    }
  }

  // out = attn_out + gate * x   (fused final add)
#pragma unroll
  for (int dt = 0; dt < 4; ++dt) {
    int c = h * HD + dt * 16 + col;
    size_t base = ((size_t)b * CDIM + c) * HWN;
    float g = gate[b * CDIM + c];
#pragma unroll
    for (int r = 0; r < 8; ++r) {
      int n = q0 + r + 8 * half;
      float v = o[dt][r] / lrun[r];
      out[base + n] = v + g * x[base + n];
    }
  }
}

// ---------------- host-side launch ----------------
extern "C" void kernel_launch(void* const* d_in, const int* in_sizes, int n_in,
                              void* d_out, int out_size, void* d_ws, size_t ws_size,
                              hipStream_t stream) {
  (void)in_sizes; (void)n_in; (void)out_size; (void)ws_size;
  const float* x    = (const float*)d_in[0];
  const float* w01  = (const float*)d_in[1];
  const float* s01  = (const float*)d_in[2];
  const float* b01  = (const float*)d_in[3];
  const float* w02  = (const float*)d_in[4];
  const float* s02  = (const float*)d_in[5];
  const float* b02  = (const float*)d_in[6];
  const float* w03  = (const float*)d_in[7];
  const float* s03  = (const float*)d_in[8];
  const float* b03  = (const float*)d_in[9];
  const float* w12  = (const float*)d_in[10];
  const float* s12  = (const float*)d_in[11];
  const float* b12  = (const float*)d_in[12];
  const float* w13  = (const float*)d_in[13];
  const float* s13  = (const float*)d_in[14];
  const float* b13  = (const float*)d_in[15];
  const float* w22  = (const float*)d_in[16];
  const float* s22  = (const float*)d_in[17];
  const float* b22  = (const float*)d_in[18];
  const float* w23  = (const float*)d_in[19];
  const float* s23  = (const float*)d_in[20];
  const float* b23  = (const float*)d_in[21];
  const float* w3   = (const float*)d_in[22];
  const float* s3   = (const float*)d_in[23];
  const float* b3   = (const float*)d_in[24];
  const float* wkv  = (const float*)d_in[25];
  const float* skv  = (const float*)d_in[26];
  const float* bkv  = (const float*)d_in[27];
  const float* wfc1 = (const float*)d_in[28];
  const float* bfc1 = (const float*)d_in[29];
  const float* wfc2 = (const float*)d_in[30];
  const float* bfc2 = (const float*)d_in[31];
  float* out = (float*)d_out;

  char* p = (char*)d_ws;
  auto alloc = [&](size_t bytes) -> char* {
    char* r = p;
    p += (bytes + 255) & ~(size_t)255;
    return r;
  };
  const size_t NX = (size_t)BSZ * CDIM * HWN;          // 16,777,216
  __bf16* xb      = (__bf16*)alloc(NX * 2);            // x in bf16
  __bf16* a01b    = (__bf16*)alloc((size_t)BSZ * 128 * HWN * 2);
  __bf16* dw0     = (__bf16*)alloc((size_t)BSZ * 128 * HWN * 2);
  __bf16* dw1     = (__bf16*)alloc((size_t)BSZ * 128 * HWN * 2);
  __bf16* dw2     = (__bf16*)alloc((size_t)BSZ * 128 * HWN * 2);
  float*  sumF    = (float*)alloc(NX * 4);             // branch sum; reused as attn
  __bf16* sumb    = (__bf16*)alloc(NX * 2);
  __bf16* pooledb = (__bf16*)alloc((size_t)BSZ * CDIM * NPOOL * 2);
  __bf16* kvb     = (__bf16*)alloc((size_t)BSZ * 1024 * NPOOL * 2);
  __bf16* w01b    = (__bf16*)alloc(128 * 512 * 2);
  __bf16* w03b    = (__bf16*)alloc(512 * 128 * 2);
  __bf16* w13b    = (__bf16*)alloc(512 * 128 * 2);
  __bf16* w23b    = (__bf16*)alloc(512 * 128 * 2);
  __bf16* w3b     = (__bf16*)alloc(512 * 512 * 2);
  __bf16* wkvb    = (__bf16*)alloc(1024 * 512 * 2);
  float*  gap     = (float*)alloc(BSZ * CDIM * 4);
  float*  y1      = (float*)alloc(BSZ * 128 * 4);
  float*  gatef   = (float*)alloc(BSZ * CDIM * 4);
  float*  attnF   = sumF;  // alias: sum is dead once sumb exists

  auto cgrid = [](size_t n) { return dim3((unsigned)((n + 255) / 256)); };

  // weight + activation converts
  cvt_f2b<<<cgrid(128 * 512), 256, 0, stream>>>(w01, w01b, 128 * 512);
  cvt_f2b<<<cgrid(512 * 128), 256, 0, stream>>>(w03, w03b, 512 * 128);
  cvt_f2b<<<cgrid(512 * 128), 256, 0, stream>>>(w13, w13b, 512 * 128);
  cvt_f2b<<<cgrid(512 * 128), 256, 0, stream>>>(w23, w23b, 512 * 128);
  cvt_f2b<<<cgrid(512 * 512), 256, 0, stream>>>(w3, w3b, 512 * 512);
  cvt_f2b<<<cgrid(1024 * 512), 256, 0, stream>>>(wkv, wkvb, 1024 * 512);
  cvt_f2b<<<cgrid(NX), 256, 0, stream>>>(x, xb, (int)NX);

  // SE gate (independent of main path; launch early)
  gap_kernel<<<BSZ * CDIM, 256, 0, stream>>>(x, gap);
  se_fc1<<<BSZ, 128, 0, stream>>>(gap, wfc1, bfc1, y1);
  se_fc2<<<BSZ, 512, 0, stream>>>(y1, wfc2, bfc2, gatef);

  // a01 = SiLU(BN(W01 @ x))   [128 x 4096] per batch
  gemm_sb_silu<EPI_WRITE_BF16><<<dim3(32, 2, BSZ), 256, 0, stream>>>(
      w01b, xb, s01, b01, nullptr, nullptr, a01b, 128, 512, HWN);

  // depthwise dilated branches
  dwconv_silu<<<cgrid((size_t)BSZ * 128 * HWN), 256, 0, stream>>>(a01b, w02, s02, b02, dw0, 3);
  dwconv_silu<<<cgrid((size_t)BSZ * 128 * HWN), 256, 0, stream>>>(a01b, w12, s12, b12, dw1, 5);
  dwconv_silu<<<cgrid((size_t)BSZ * 128 * HWN), 256, 0, stream>>>(a01b, w22, s22, b22, dw2, 7);

  // three 128->512 pointwise convs, SiLU'd then summed into sumF
  gemm_sb_silu<EPI_ACC_F32 | EPI_FIRST><<<dim3(32, 8, BSZ), 256, 0, stream>>>(
      w03b, dw0, s03, b03, nullptr, sumF, nullptr, 512, 128, HWN);
  gemm_sb_silu<EPI_ACC_F32><<<dim3(32, 8, BSZ), 256, 0, stream>>>(
      w13b, dw1, s13, b13, nullptr, sumF, nullptr, 512, 128, HWN);
  gemm_sb_silu<EPI_ACC_F32><<<dim3(32, 8, BSZ), 256, 0, stream>>>(
      w23b, dw2, s23, b23, nullptr, sumF, nullptr, 512, 128, HWN);
  cvt_f2b<<<cgrid(NX), 256, 0, stream>>>(sumF, sumb, (int)NX);

  // attn = SiLU(BN(W3 @ sum)) * x     (writes attnF, aliasing sumF)
  gemm_sb_silu<EPI_WRITE_F32 | EPI_MULX><<<dim3(32, 8, BSZ), 256, 0, stream>>>(
      w3b, sumb, s3, b3, x, attnF, nullptr, 512, 512, HWN);

  // pool to 16x16 -> bf16, then kv = SiLU(BN(Wkv @ pooled))  [1024 x 256]
  pool_kernel<<<cgrid((size_t)BSZ * CDIM * NPOOL), 256, 0, stream>>>(attnF, pooledb);
  gemm_sb_silu<EPI_WRITE_BF16><<<dim3(2, 16, BSZ), 256, 0, stream>>>(
      wkvb, pooledb, skv, bkv, nullptr, nullptr, kvb, 1024, 512, NPOOL);

  // fused pooled cross-attention + gate*x epilogue -> d_out
  attention_kernel<<<dim3(64, NHEAD, BSZ), 128, 0, stream>>>(xb, kvb, x, gatef, out);
}
